// AnticipatoryRestaurantGNN_32933809226525
// MI455X (gfx1250) — compile-verified, matrix operand order checked by reading, not run on hardware
//
#include <hip/hip_runtime.h>
#include <hip/hip_bf16.h>
#include <math.h>

#define HIDN   256
#define HEADS  4
#define LAYERS 4
#define INDIM  64
#define EDIM   16
#define NGRP   64
#define EPSBN  1e-5f

typedef __attribute__((ext_vector_type(16))) __bf16 v16bf;
typedef __attribute__((ext_vector_type(8)))  __bf16 v8bf;
typedef __attribute__((ext_vector_type(8)))  float  v8f;

// ---------------------------------------------------------------- fills
__global__ void fill_f32(float* p, float v, int n) {
    int i = blockIdx.x * blockDim.x + threadIdx.x;
    if (i < n) p[i] = v;
}
__global__ void fill_i32(int* p, int v, int n) {
    int i = blockIdx.x * blockDim.x + threadIdx.x;
    if (i < n) p[i] = v;
}

// ---------------------------------------------------------------- converts
// f32 -> bf16, 8 elements per thread (n must be a multiple of 8)
__global__ void cvt_bf16(const float* __restrict__ in, __bf16* __restrict__ outp, long long n) {
    long long base = ((long long)blockIdx.x * blockDim.x + threadIdx.x) * 8;
    if (base >= n) return;
    const float4* s = (const float4*)(in + base);
    float4 a = s[0], b = s[1];
    __bf16* d = outp + base;
    d[0] = (__bf16)a.x; d[1] = (__bf16)a.y; d[2] = (__bf16)a.z; d[3] = (__bf16)a.w;
    d[4] = (__bf16)b.x; d[5] = (__bf16)b.y; d[6] = (__bf16)b.z; d[7] = (__bf16)b.w;
}
// W[K][Nout] f32 -> Wt[Nout][K] bf16 (transpose + convert)
__global__ void cvtT_bf16(const float* __restrict__ in, __bf16* __restrict__ outp, int K, int Nout) {
    int i = blockIdx.x * blockDim.x + threadIdx.x;
    if (i >= K * Nout) return;
    int k = i % K;
    int n = i / K;
    outp[(size_t)n * K + k] = (__bf16)in[(size_t)k * Nout + n];
}

// ---------------------------------------------------------------- async helpers
__device__ __forceinline__ void async_ld128(unsigned int lds_byte, unsigned long long gaddr) {
    asm volatile("global_load_async_to_lds_b128 %0, %1, off"
                 :: "v"(lds_byte), "v"(gaddr) : "memory");
}
__device__ __forceinline__ void wait_async3() {
    asm volatile("s_wait_asynccnt 0x3" ::: "memory");
}
__device__ __forceinline__ void wait_async0() {
    asm volatile("s_wait_asynccnt 0x0" ::: "memory");
}

// ---------------------------------------------------------------- WMMA GEMM (bf16 in, f32 out)
// C[M,Nout] = A[M,K](bf16) @ Wt[Nout,K](bf16, pre-transposed) + bias
// block = 256 threads (8 waves), tile 64(M) x 128(N), K-step 32,
// double-buffered LDS filled by global_load_async_to_lds_b128 (ASYNCcnt pipelined).
__global__ __launch_bounds__(256) void wmma_gemm_async(
    const __bf16* __restrict__ A, const __bf16* __restrict__ Wt,
    const float* __restrict__ bias, float* __restrict__ C,
    int M, int K, int Nout)
{
    __shared__ __attribute__((aligned(16))) __bf16 lA[2][64 * 32];   // [buf][row][k]
    __shared__ __attribute__((aligned(16))) __bf16 lB[2][128 * 32];  // [buf][n][k]

    const int tid  = threadIdx.x;
    const int m0   = blockIdx.x * 64;
    const int n0   = blockIdx.y * 128;
    const int wid  = tid >> 5;
    const int lane = tid & 31;
    const int r0   = (wid >> 2) * 32;
    const int c0   = (wid & 3) * 32;

    v8f acc[2][2] = {};

    // async copy mapping: A tile = 64 rows x 64B, one 16B chunk per thread
    const int arow = tid >> 2;
    const int ae   = (tid & 3) * 8;         // element offset within 32-k row
    int growA = m0 + arow;
    if (growA >= M) growA = M - 1;          // clamp: no EXEC divergence, junk rows unstored
    // B tile = 128 rows x 64B, two 16B chunks per thread
    const int bn0 = tid >> 1;               // chunks tid -> (n = tid>>2 ... ) use two ids:
    (void)bn0;

    const int nsteps = K >> 5;

#define ISSUE_TILE(buf, kk)                                                              \
    do {                                                                                 \
        unsigned int la = (unsigned int)(size_t)(&lA[(buf)][arow * 32 + ae]);            \
        unsigned long long ga = (unsigned long long)(size_t)(A + (size_t)growA * K + (kk) + ae); \
        async_ld128(la, ga);                                                             \
        int c_0 = tid;        int nb0 = c_0 >> 2; int eb0 = (c_0 & 3) * 8;               \
        unsigned int lb0 = (unsigned int)(size_t)(&lB[(buf)][nb0 * 32 + eb0]);           \
        unsigned long long gb0 = (unsigned long long)(size_t)(Wt + (size_t)(n0 + nb0) * K + (kk) + eb0); \
        async_ld128(lb0, gb0);                                                           \
        int c_1 = tid + 256;  int nb1 = c_1 >> 2; int eb1 = (c_1 & 3) * 8;               \
        unsigned int lb1 = (unsigned int)(size_t)(&lB[(buf)][nb1 * 32 + eb1]);           \
        unsigned long long gb1 = (unsigned long long)(size_t)(Wt + (size_t)(n0 + nb1) * K + (kk) + eb1); \
        async_ld128(lb1, gb1);                                                           \
    } while (0)

    ISSUE_TILE(0, 0);

    for (int s = 0; s < nsteps; ++s) {
        const int buf = s & 1;
        if (s + 1 < nsteps) {
            ISSUE_TILE(buf ^ 1, (s + 1) << 5);   // copy for next step overlaps this step's WMMA
            wait_async3();                        // own 3 loads for current buffer complete
        } else {
            wait_async0();
        }
        __syncthreads();

        // fragment loads per documented 16-bit A layout:
        // lanes 0-15: K = {0..7, 16..23}; lanes 16-31: K = {8..15, 24..31}
        const int ks   = (lane >> 4) * 8;
        const int mloc = lane & 15;
        v16bf afr[2], bfr[2];
        #pragma unroll
        for (int t = 0; t < 2; ++t) {
            const __bf16* pa = &lA[buf][(r0 + t * 16 + mloc) * 32 + ks];
            v8bf alo = *(const v8bf*)pa;
            v8bf ahi = *(const v8bf*)(pa + 16);
            afr[t] = __builtin_shufflevector(alo, ahi, 0,1,2,3,4,5,6,7,8,9,10,11,12,13,14,15);
            const __bf16* pb = &lB[buf][(c0 + t * 16 + mloc) * 32 + ks];
            v8bf blo = *(const v8bf*)pb;
            v8bf bhi = *(const v8bf*)(pb + 16);
            bfr[t] = __builtin_shufflevector(blo, bhi, 0,1,2,3,4,5,6,7,8,9,10,11,12,13,14,15);
        }
        #pragma unroll
        for (int si = 0; si < 2; ++si)
            #pragma unroll
            for (int sj = 0; sj < 2; ++sj)
                acc[si][sj] = __builtin_amdgcn_wmma_f32_16x16x32_bf16(
                    false, afr[si], false, bfr[sj], (short)0, acc[si][sj], false, false);
        __syncthreads();   // all waves done reading this buffer before it is refilled
    }
#undef ISSUE_TILE

    // store per documented 16x16 f32 C/D layout
    const int nl = lane & 15;
    const int mo = (lane >> 4) * 8;
    #pragma unroll
    for (int si = 0; si < 2; ++si) {
        #pragma unroll
        for (int sj = 0; sj < 2; ++sj) {
            int n = n0 + c0 + sj * 16 + nl;
            float b = bias[n];
            int mb = m0 + r0 + si * 16 + mo;
            #pragma unroll
            for (int j = 0; j < 8; ++j) {
                int m = mb + j;
                if (m < M) C[(size_t)m * Nout + n] = acc[si][sj][j] + b;
            }
        }
    }
}

// ---------------------------------------------------------------- edge phase
__device__ __forceinline__ int enc_f32(float x) {
    int i = __float_as_int(x);
    return i >= 0 ? i : (i ^ 0x7fffffff);
}
__device__ __forceinline__ float dec_f32(int i) {
    return __int_as_float(i >= 0 ? i : (i ^ 0x7fffffff));
}

// wave-per-edge: alpha logit = (q[dst] . (k[src]+e)) / sqrt(64); atomicMax into amax
__global__ __launch_bounds__(256) void edge_logits(
    const float* __restrict__ q, const float* __restrict__ k,
    const float* __restrict__ ea, const float* __restrict__ We,
    const float* __restrict__ be,
    const int* __restrict__ srcI, const int* __restrict__ dstI,
    float* __restrict__ alpha, int* __restrict__ amax, int E)
{
    int w    = (blockIdx.x * blockDim.x + threadIdx.x) >> 5;
    int lane = threadIdx.x & 31;
    if (w >= E) return;
    int s = srcI[w], d = dstI[w];
    int ch0 = lane * 8;
    __builtin_prefetch(&q[(size_t)d * HIDN + ch0], 0, 3);   // L2-resident gather rows
    __builtin_prefetch(&k[(size_t)s * HIDN + ch0], 0, 3);
    float e16[EDIM];
    #pragma unroll
    for (int t = 0; t < EDIM; ++t) e16[t] = ea[(size_t)w * EDIM + t];
    float partial = 0.f;
    #pragma unroll
    for (int j = 0; j < 8; ++j) {
        int ch = ch0 + j;
        float ev = be[ch];
        #pragma unroll
        for (int t = 0; t < EDIM; ++t) ev = fmaf(e16[t], We[t * HIDN + ch], ev);
        partial = fmaf(q[(size_t)d * HIDN + ch], k[(size_t)s * HIDN + ch] + ev, partial);
    }
    for (int o = 1; o < 8; o <<= 1) partial += __shfl_xor(partial, o, 32);
    if ((lane & 7) == 0) {
        int h = lane >> 3;
        float logit = partial * 0.125f; // 1/sqrt(C=64)
        alpha[(size_t)w * HEADS + h] = logit;
        atomicMax(&amax[d * HEADS + h], enc_f32(logit));
    }
}

__global__ void edge_softmax(
    const int* __restrict__ dstI, const int* __restrict__ amax,
    float* __restrict__ alpha, float* __restrict__ denom, int EH)
{
    int i = blockIdx.x * blockDim.x + threadIdx.x;
    if (i >= EH) return;
    int e = i >> 2, h = i & 3;
    int d = dstI[e];
    float mx = dec_f32(amax[d * HEADS + h]);
    float wexp = __expf(alpha[i] - mx);
    alpha[i] = wexp;
    atomicAdd(&denom[d * HEADS + h], wexp);
}

// wave-per-edge: out[dst] += (alpha/denom) * (v[src] + e)   (e recomputed)
__global__ __launch_bounds__(256) void edge_aggregate(
    const float* __restrict__ v, const float* __restrict__ ea,
    const float* __restrict__ We, const float* __restrict__ be,
    const int* __restrict__ srcI, const int* __restrict__ dstI,
    const float* __restrict__ alpha, const float* __restrict__ denom,
    float* __restrict__ out, int E)
{
    int w    = (blockIdx.x * blockDim.x + threadIdx.x) >> 5;
    int lane = threadIdx.x & 31;
    if (w >= E) return;
    int s = srcI[w], d = dstI[w];
    int ch0 = lane * 8;
    __builtin_prefetch(&v[(size_t)s * HIDN + ch0], 0, 3);
    int h = lane >> 3;
    float wgt = alpha[(size_t)w * HEADS + h] / denom[d * HEADS + h];
    float e16[EDIM];
    #pragma unroll
    for (int t = 0; t < EDIM; ++t) e16[t] = ea[(size_t)w * EDIM + t];
    #pragma unroll
    for (int j = 0; j < 8; ++j) {
        int ch = ch0 + j;
        float ev = be[ch];
        #pragma unroll
        for (int t = 0; t < EDIM; ++t) ev = fmaf(e16[t], We[t * HIDN + ch], ev);
        atomicAdd(&out[(size_t)d * HIDN + ch], wgt * (v[(size_t)s * HIDN + ch] + ev));
    }
}

// ---------------------------------------------------------------- beta gate + BN stats
__global__ __launch_bounds__(256) void beta_combine(
    const float* __restrict__ out, const float* __restrict__ xr,
    const float* __restrict__ Wb, float* __restrict__ x,
    float* __restrict__ bn_sum, float* __restrict__ bn_sumsq, int N)
{
    __shared__ float s1[HIDN];
    __shared__ float s2[HIDN];
    int tid = threadIdx.x;
    if (tid < HIDN) { s1[tid] = 0.f; s2[tid] = 0.f; }
    __syncthreads();

    int node = blockIdx.x * 8 + (tid >> 5);
    int lane = tid & 31;
    bool active = node < N;
    float o8[8] = {0}, r8[8] = {0};
    float z = 0.f;
    int ch0 = lane * 8;
    if (active) {
        #pragma unroll
        for (int j = 0; j < 8; ++j) {
            int ch = ch0 + j;
            float o = out[(size_t)node * HIDN + ch];
            float r = xr[(size_t)node * HIDN + ch];
            o8[j] = o; r8[j] = r;
            z = fmaf(o, Wb[ch], z);
            z = fmaf(r, Wb[HIDN + ch], z);
            z = fmaf(o - r, Wb[2 * HIDN + ch], z);
        }
    }
    for (int o = 1; o < 32; o <<= 1) z += __shfl_xor(z, o, 32);
    if (active) {
        float beta = 1.f / (1.f + __expf(-z));
        #pragma unroll
        for (int j = 0; j < 8; ++j) {
            int ch = ch0 + j;
            float xn = beta * r8[j] + (1.f - beta) * o8[j];
            x[(size_t)node * HIDN + ch] = xn;
            atomicAdd(&s1[ch], xn);          // LDS ds_add_f32
            atomicAdd(&s2[ch], xn * xn);
        }
    }
    __syncthreads();
    if (tid < HIDN) {
        atomicAdd(&bn_sum[tid],   s1[tid]);
        atomicAdd(&bn_sumsq[tid], s2[tid]);
    }
}

__global__ void bn_leaky(
    float* __restrict__ x, const float* __restrict__ bn_sum,
    const float* __restrict__ bn_sumsq, const float* __restrict__ gamma,
    const float* __restrict__ bbeta, int total, float invN)
{
    int i = blockIdx.x * blockDim.x + threadIdx.x;
    if (i >= total) return;
    int ch = i & (HIDN - 1);
    float mu  = bn_sum[ch] * invN;
    float var = bn_sumsq[ch] * invN - mu * mu;
    float y = (x[i] - mu) * rsqrtf(var + EPSBN) * gamma[ch] + bbeta[ch];
    x[i] = y >= 0.f ? y : 0.1f * y;
}

// ---------------------------------------------------------------- pooling + head
__global__ void pool_accum(
    const float* __restrict__ x, const int* __restrict__ batch,
    float* __restrict__ pooled, float* __restrict__ counts, int total)
{
    int i = blockIdx.x * blockDim.x + threadIdx.x;
    if (i >= total) return;
    int n = i >> 8, ch = i & (HIDN - 1);
    int g = batch[n];
    atomicAdd(&pooled[(size_t)g * HIDN + ch], x[i]);
    if (ch == 0) atomicAdd(&counts[g], 1.f);
}

__global__ void head_kernel(
    const float* __restrict__ pooled, const float* __restrict__ counts,
    const float* __restrict__ hw, const float* __restrict__ hb,
    float* __restrict__ y, int G)
{
    int g = blockIdx.x * blockDim.x + threadIdx.x;
    if (g >= G) return;
    float c = fmaxf(counts[g], 1.f);
    float acc = hb[0];
    for (int ch = 0; ch < HIDN; ++ch) {
        float sm = pooled[(size_t)g * HIDN + ch];
        acc = fmaf(sm / c, hw[ch], acc);
        acc = fmaf(sm, hw[HIDN + ch], acc);
    }
    y[g] = acc;
}

// ---------------------------------------------------------------- launcher
extern "C" void kernel_launch(void* const* d_in, const int* in_sizes, int n_in,
                              void* d_out, int out_size, void* d_ws, size_t ws_size,
                              hipStream_t stream)
{
    const float* node_features = (const float*)d_in[0];
    const int*   edge_index    = (const int*)d_in[1];
    const float* edge_attr     = (const float*)d_in[2];
    const int*   batch         = (const int*)d_in[3];
    const float* proj_W        = (const float*)d_in[4];
    const float* proj_b        = (const float*)d_in[5];
    const float* Wq            = (const float*)d_in[6];
    const float* bq            = (const float*)d_in[7];
    const float* Wk            = (const float*)d_in[8];
    const float* bk            = (const float*)d_in[9];
    const float* Wv            = (const float*)d_in[10];
    const float* bv            = (const float*)d_in[11];
    const float* We            = (const float*)d_in[12];
    const float* be            = (const float*)d_in[13];
    const float* Wskip         = (const float*)d_in[14];
    const float* bskip         = (const float*)d_in[15];
    const float* Wbeta         = (const float*)d_in[16];
    const float* bn_gamma      = (const float*)d_in[17];
    const float* bn_beta       = (const float*)d_in[18];
    const float* head_W        = (const float*)d_in[19];
    const float* head_b        = (const float*)d_in[20];

    const int N = in_sizes[0] / INDIM;
    const int E = in_sizes[1] / 2;
    const int* srcI = edge_index;
    const int* dstI = edge_index + E;

    char* ws = (char*)d_ws;
    size_t off = 0;
    auto alloc = [&](size_t bytes) -> char* {
        char* p = ws + off;
        off += (bytes + 255) & ~(size_t)255;
        return p;
    };
    float*  x     = (float*) alloc((size_t)N * HIDN * 4);
    float*  q     = (float*) alloc((size_t)N * HIDN * 4);
    float*  k     = (float*) alloc((size_t)N * HIDN * 4);
    float*  v     = (float*) alloc((size_t)N * HIDN * 4);
    float*  xr    = (float*) alloc((size_t)N * HIDN * 4);
    float*  outb  = (float*) alloc((size_t)N * HIDN * 4);
    float*  alpha = (float*) alloc((size_t)E * HEADS * 4);
    int*    amax  = (int*)   alloc((size_t)N * HEADS * 4);
    float*  denom = (float*) alloc((size_t)N * HEADS * 4);
    float*  bns   = (float*) alloc(2 * HIDN * 4);
    float*  pooled= (float*) alloc((size_t)NGRP * HIDN * 4);
    float*  counts= (float*) alloc(NGRP * 4);
    __bf16* xbf   = (__bf16*)alloc((size_t)N * HIDN * 2);   // bf16 activations
    __bf16* wtbf  = (__bf16*)alloc((size_t)HIDN * HIDN * 2);// bf16 transposed weight scratch

    const int mTiles = (N + 63) / 64;
    dim3 gemmGrid(mTiles, HIDN / 128);
    const int edgeBlocks = (E + 7) / 8;
    const int nodeBlocks = (N + 7) / 8;
    const int ehBlocks   = (E * HEADS + 255) / 256;
    const int total      = N * HIDN;
    const int elemBlocks = (total + 255) / 256;

    // input projection: x = node_features @ proj_W + proj_b (bf16 WMMA path)
    {
        long long nconv = (long long)N * INDIM;
        cvt_bf16<<<(int)((nconv / 8 + 255) / 256), 256, 0, stream>>>(node_features, xbf, nconv);
        cvtT_bf16<<<(INDIM * HIDN + 255) / 256, 256, 0, stream>>>(proj_W, wtbf, INDIM, HIDN);
        wmma_gemm_async<<<gemmGrid, 256, 0, stream>>>(xbf, wtbf, proj_b, x, N, INDIM, HIDN);
    }

    for (int l = 0; l < LAYERS; ++l) {
        const float* Wql = Wq + (size_t)l * HIDN * HIDN;
        const float* Wkl = Wk + (size_t)l * HIDN * HIDN;
        const float* Wvl = Wv + (size_t)l * HIDN * HIDN;
        const float* Wsl = Wskip + (size_t)l * HIDN * HIDN;
        const float* Wel = We + (size_t)l * EDIM * HIDN;
        const float* Wbl = Wbeta + (size_t)l * 3 * HIDN;

        // init accumulators
        fill_i32<<<(N * HEADS + 255) / 256, 256, 0, stream>>>(amax, (int)0x80000000, N * HEADS);
        fill_f32<<<(N * HEADS + 255) / 256, 256, 0, stream>>>(denom, 0.f, N * HEADS);
        fill_f32<<<elemBlocks, 256, 0, stream>>>(outb, 0.f, total);
        fill_f32<<<(2 * HIDN + 255) / 256, 256, 0, stream>>>(bns, 0.f, 2 * HIDN);

        // x -> bf16 once, reused by all 4 projections
        long long nconv = (long long)N * HIDN;
        cvt_bf16<<<(int)((nconv / 8 + 255) / 256), 256, 0, stream>>>(x, xbf, nconv);

        cvtT_bf16<<<(HIDN * HIDN + 255) / 256, 256, 0, stream>>>(Wql, wtbf, HIDN, HIDN);
        wmma_gemm_async<<<gemmGrid, 256, 0, stream>>>(xbf, wtbf, bq + l * HIDN, q, N, HIDN, HIDN);
        cvtT_bf16<<<(HIDN * HIDN + 255) / 256, 256, 0, stream>>>(Wkl, wtbf, HIDN, HIDN);
        wmma_gemm_async<<<gemmGrid, 256, 0, stream>>>(xbf, wtbf, bk + l * HIDN, k, N, HIDN, HIDN);
        cvtT_bf16<<<(HIDN * HIDN + 255) / 256, 256, 0, stream>>>(Wvl, wtbf, HIDN, HIDN);
        wmma_gemm_async<<<gemmGrid, 256, 0, stream>>>(xbf, wtbf, bv + l * HIDN, v, N, HIDN, HIDN);
        cvtT_bf16<<<(HIDN * HIDN + 255) / 256, 256, 0, stream>>>(Wsl, wtbf, HIDN, HIDN);
        wmma_gemm_async<<<gemmGrid, 256, 0, stream>>>(xbf, wtbf, bskip + l * HIDN, xr, N, HIDN, HIDN);

        // edge attention
        edge_logits<<<edgeBlocks, 256, 0, stream>>>(q, k, edge_attr, Wel, be + l * HIDN,
                                                    srcI, dstI, alpha, amax, E);
        edge_softmax<<<ehBlocks, 256, 0, stream>>>(dstI, amax, alpha, denom, E * HEADS);
        edge_aggregate<<<edgeBlocks, 256, 0, stream>>>(v, edge_attr, Wel, be + l * HIDN,
                                                       srcI, dstI, alpha, denom, outb, E);

        // gate + BN stats, then normalize + leaky relu (in place on x)
        beta_combine<<<nodeBlocks, 256, 0, stream>>>(outb, xr, Wbl, x, bns, bns + HIDN, N);
        bn_leaky<<<elemBlocks, 256, 0, stream>>>(x, bns, bns + HIDN,
                                                 bn_gamma + l * HIDN, bn_beta + l * HIDN,
                                                 total, 1.0f / (float)N);
    }

    // pooling + head
    fill_f32<<<(NGRP * HIDN + 255) / 256, 256, 0, stream>>>(pooled, 0.f, NGRP * HIDN);
    fill_f32<<<1, 256, 0, stream>>>(counts, 0.f, NGRP);
    pool_accum<<<elemBlocks, 256, 0, stream>>>(x, batch, pooled, counts, total);
    head_kernel<<<1, NGRP, 0, stream>>>(pooled, counts, head_W, head_b, (float*)d_out, NGRP);
}
